// CondSdpaAttention_42752104464452
// MI455X (gfx1250) — compile-verified, matrix operand order checked by reading, not run on hardware
//
#include <hip/hip_runtime.h>

// ---------------------------------------------------------------------------
// CDNA5 (gfx1250) fused attention block, bf16 WMMA pipeline.
// B=4, S=2048, H=2048, C=512, QK=2560, NH=16, dqk=160, dv=128
// ---------------------------------------------------------------------------

typedef __attribute__((ext_vector_type(16))) __bf16 v16bf;
typedef __attribute__((ext_vector_type(8)))  float  v8f;
typedef unsigned int u32x4 __attribute__((ext_vector_type(4)));
typedef int          i32x4 __attribute__((ext_vector_type(4)));
typedef int          i32x8 __attribute__((ext_vector_type(8)));

// async-copy pointer types: (global int4*, lds int4*)
typedef int v4i __attribute__((ext_vector_type(4)));
typedef __attribute__((address_space(1))) v4i gv4i;
typedef __attribute__((address_space(3))) v4i lv4i;

union Frag16 { v16bf v; uint4 q[2]; };

#if __has_builtin(__builtin_amdgcn_global_load_async_to_lds_b128)
#define HAVE_ASYNC_B128 1
#endif
#if __has_builtin(__builtin_amdgcn_tensor_load_to_lds)
#define HAVE_TDM 1
#endif

static __device__ __forceinline__ void wait_async0() {
#if __has_builtin(__builtin_amdgcn_s_wait_asynccnt)
    __builtin_amdgcn_s_wait_asynccnt(0);
#else
    asm volatile("s_wait_asynccnt 0x0" ::: "memory");
#endif
}

static __device__ __forceinline__ void wait_tensor0() {
#if __has_builtin(__builtin_amdgcn_s_wait_tensorcnt)
    __builtin_amdgcn_s_wait_tensorcnt(0);
#else
    asm volatile("s_wait_tensorcnt 0x0" ::: "memory");
#endif
}

static __device__ __forceinline__ __bf16 f2bf(float f) {
    unsigned u = __builtin_bit_cast(unsigned, f);
    unsigned r = u + 0x7fffu + ((u >> 16) & 1u);
    unsigned short h = (unsigned short)(r >> 16);
    return __builtin_bit_cast(__bf16, h);
}

// ---------------------------------------------------------------------------
// Generic bf16 GEMM:  C[M,N] (fp32) = A[M,K] (bf16, row-major) @ W[N,K]^T
// Block = 256 threads (8 waves), tile 128x128, K panel 64.
// Wave grid 2x4: each wave owns a 64x32 sub-tile = 4x2 WMMA accumulators.
// ---------------------------------------------------------------------------
#define GBM 128
#define GBN 128
#define GKB 64

__global__ __launch_bounds__(256)
void gemm_bf16_f32(const __bf16* __restrict__ A, const __bf16* __restrict__ W,
                   float* __restrict__ C, int M, int N, int K) {
    __shared__ __align__(16) __bf16 As[GBM][GKB];
    __shared__ __align__(16) __bf16 Bs[GBN][GKB];

    const int tid  = threadIdx.x;
    const int wave = tid >> 5;
    const int lane = tid & 31;
    const int lrow = lane & 15;          // M-row (A) / N-col (B) inside 16
    const int hi   = lane >> 4;          // lane half
    const int kb8  = hi * 8;             // K sub-base per ISA striping

    const int bm0 = blockIdx.y * GBM;
    const int bn0 = blockIdx.x * GBN;
    const int wm0 = (wave >> 2) * 64;    // wave M offset in block
    const int wn0 = (wave & 3) * 32;     // wave N offset in block

    v8f acc[4][2];
#pragma unroll
    for (int i = 0; i < 4; ++i)
#pragma unroll
        for (int j = 0; j < 2; ++j) acc[i][j] = (v8f)(0.0f);

    for (int k0 = 0; k0 < K; k0 += GKB) {
        // ---- stage 128x64 A/B panels into LDS ----
#if defined(HAVE_ASYNC_B128)
#pragma unroll
        for (int it = 0; it < 4; ++it) {
            int idx = tid + it * 256;         // 0..1023
            int r   = idx >> 3;               // 0..127
            int c8  = (idx & 7) * 8;          // 0..56
            __builtin_amdgcn_global_load_async_to_lds_b128(
                (gv4i*)&A[(size_t)(bm0 + r) * K + k0 + c8],
                (lv4i*)&As[r][c8], 0, 0);
            __builtin_amdgcn_global_load_async_to_lds_b128(
                (gv4i*)&W[(size_t)(bn0 + r) * K + k0 + c8],
                (lv4i*)&Bs[r][c8], 0, 0);
        }
        wait_async0();
#else
#pragma unroll
        for (int it = 0; it < 4; ++it) {
            int idx = tid + it * 256;
            int r   = idx >> 3;
            int c8  = (idx & 7) * 8;
            *(uint4*)&As[r][c8] = *(const uint4*)&A[(size_t)(bm0 + r) * K + k0 + c8];
            *(uint4*)&Bs[r][c8] = *(const uint4*)&W[(size_t)(bn0 + r) * K + k0 + c8];
            if (k0 + GKB < K) {
                __builtin_prefetch(&A[(size_t)(bm0 + r) * K + k0 + GKB + c8], 0, 1);
                __builtin_prefetch(&W[(size_t)(bn0 + r) * K + k0 + GKB + c8], 0, 1);
            }
        }
#endif
        __syncthreads();

#pragma unroll
        for (int kk = 0; kk < GKB; kk += 32) {
            // Batch all fragment loads first so WMMAs issue back-to-back.
            Frag16 bf[2], af[4];
#pragma unroll
            for (int ni = 0; ni < 2; ++ni) {
                const __bf16* bp = &Bs[wn0 + ni * 16 + lrow][kk + kb8];
                bf[ni].q[0] = *(const uint4*)(bp);
                bf[ni].q[1] = *(const uint4*)(bp + 16);
            }
#pragma unroll
            for (int mi = 0; mi < 4; ++mi) {
                const __bf16* ap = &As[wm0 + mi * 16 + lrow][kk + kb8];
                af[mi].q[0] = *(const uint4*)(ap);
                af[mi].q[1] = *(const uint4*)(ap + 16);
            }
#pragma unroll
            for (int mi = 0; mi < 4; ++mi)
#pragma unroll
                for (int ni = 0; ni < 2; ++ni)
                    acc[mi][ni] = __builtin_amdgcn_wmma_f32_16x16x32_bf16(
                        false, af[mi].v, false, bf[ni].v, (short)0, acc[mi][ni],
                        false, false);
        }
        __syncthreads();
    }

    // Epilogue: C layout — VGPR r holds row (r + 8*hi), lane lrow = column.
#pragma unroll
    for (int mi = 0; mi < 4; ++mi) {
#pragma unroll
        for (int ni = 0; ni < 2; ++ni) {
            int col = bn0 + wn0 + ni * 16 + lrow;
#pragma unroll
            for (int r = 0; r < 8; ++r) {
                int row = bm0 + wm0 + mi * 16 + r + 8 * hi;
                C[(size_t)row * N + col] = acc[mi][ni][r];
            }
        }
    }
}

// ---------------------------------------------------------------------------
// Pack: qk_in = concat(hidden, cond) -> bf16 [B*S, 2560]; xh = bf16 hidden.
// ---------------------------------------------------------------------------
__global__ void pack_inputs(const float* __restrict__ hid, const float* __restrict__ cond,
                            __bf16* __restrict__ qkin, __bf16* __restrict__ xh,
                            size_t total) {
    size_t i = (size_t)blockIdx.x * blockDim.x + threadIdx.x;
    if (i >= total) return;
    size_t row = i / 2560, col = i % 2560;
    float v;
    if (col < 2048) {
        v = hid[row * 2048 + col];
        xh[row * 2048 + col] = f2bf(v);
    } else {
        v = cond[row * 512 + (col - 2048)];
    }
    qkin[i] = f2bf(v);
}

__global__ void conv_f32_bf16(const float* __restrict__ src, __bf16* __restrict__ dst,
                              size_t n) {
    size_t i = (size_t)blockIdx.x * blockDim.x + threadIdx.x;
    if (i < n) dst[i] = f2bf(src[i]);
}

// ---------------------------------------------------------------------------
// RoPE + head-major repack: X fp32 [B*S,2560] -> Out bf16 [B,16,S,160]
// ---------------------------------------------------------------------------
__global__ void rope_repack(const float* __restrict__ X, const float* __restrict__ cs,
                            const float* __restrict__ sn, __bf16* __restrict__ Out,
                            int S, size_t total) {
    size_t i = (size_t)blockIdx.x * blockDim.x + threadIdx.x;
    if (i >= total) return;
    size_t row = i / 2560;                 // b*S + s
    int col = (int)(i % 2560);
    int b = (int)(row / S), s = (int)(row % S);
    int h = col / 160, dd = col % 160;
    float x = X[i];
    float rot = (dd < 80) ? -X[row * 2560 + h * 160 + dd + 80]
                          :  X[row * 2560 + h * 160 + dd - 80];
    float val = x * cs[(size_t)s * 160 + dd] + rot * sn[(size_t)s * 160 + dd];
    Out[(((size_t)b * 16 + h) * S + s) * 160 + dd] = f2bf(val);
}

// V repack: vf32 [B*S,2048] -> bf16 [B,16,S,128]
__global__ void v_repack(const float* __restrict__ X, __bf16* __restrict__ Out,
                         int S, size_t total) {
    size_t i = (size_t)blockIdx.x * blockDim.x + threadIdx.x;
    if (i >= total) return;
    size_t row = i / 2048;
    int col = (int)(i % 2048);
    int b = (int)(row / S), s = (int)(row % S);
    int h = col >> 7, d = col & 127;
    Out[(((size_t)b * 16 + h) * S + s) * 128 + d] = f2bf(X[i]);
}

// ---------------------------------------------------------------------------
// Flash attention core. Block = 128 threads (4 waves), 64 q-rows per block
// (16 per wave), keys staged 64 at a time through LDS (K tile via TDM when
// available). dqk=160, dv=128.
// Q/K: bf16 [B,16,S,160]  V: bf16 [B,16,S,128]  O: bf16 [B*S, 2048]
// ---------------------------------------------------------------------------
#define AQK 64
__global__ __launch_bounds__(128)
void attn_flash(const __bf16* __restrict__ Q, const __bf16* __restrict__ Kg,
                const __bf16* __restrict__ V, __bf16* __restrict__ O, int S) {
    __shared__ __align__(16) __bf16 Ks[AQK][160];   // 20 KB, key-major
    __shared__ __align__(16) __bf16 Vs[128][AQK];   // 16 KB, d-major (transposed)
    __shared__ __align__(16) __bf16 Ps[4][16][32];  // 4 KB, per-wave P tile

    const int b = blockIdx.z, h = blockIdx.y;
    const int q0 = blockIdx.x * 64;
    const int tid  = threadIdx.x;
    const int wave = tid >> 5;
    const int lane = tid & 31;
    const int lrow = lane & 15;
    const int hi   = lane >> 4;
    const int kb8  = hi * 8;

    const size_t baseQK = ((size_t)(b * 16 + h) * S) * 160;
    const size_t baseV  = ((size_t)(b * 16 + h) * S) * 128;

    // Persistent A-fragments of this wave's 16 q rows (K = 160 -> 5 frags).
    Frag16 qf[5];
    {
        const __bf16* qr = Q + baseQK + (size_t)(q0 + wave * 16 + lrow) * 160;
#pragma unroll
        for (int d0 = 0; d0 < 5; ++d0) {
            qf[d0].q[0] = *(const uint4*)(qr + d0 * 32 + kb8);
            qf[d0].q[1] = *(const uint4*)(qr + d0 * 32 + kb8 + 16);
        }
    }

    v8f acc[8];
#pragma unroll
    for (int j = 0; j < 8; ++j) acc[j] = (v8f)(0.0f);
    float mrow[8], lsum[8];
#pragma unroll
    for (int r = 0; r < 8; ++r) { mrow[r] = -1.0e30f; lsum[r] = 0.0f; }

    const float scale = 0.07905694150420949f;        // 1/sqrt(160)
    const float LOG2E = 1.4426950408889634f;

    for (int k0 = 0; k0 < S; k0 += AQK) {
        // ---- stage K tile (key-major): TDM when available ----
#if defined(HAVE_TDM)
        if (tid < 32) {   // single wave issues the DMA (uniform per wave)
            unsigned long long ga =
                (unsigned long long)(uintptr_t)(Kg + baseQK + (size_t)k0 * 160);
            unsigned lds_addr = (unsigned)(uintptr_t)(&Ks[0][0]);
            // D# group0: count=1 | lds_addr | global_addr | type=2
            u32x4 g0 = { 1u, lds_addr, (unsigned)ga,
                         (unsigned)((ga >> 32) & 0x1ffffffu) | (2u << 30) };
            // D# group1: data_size=2B, dim0=160, dim1=S, tile 160x64, stride0=160
            i32x8 g1 = { (int)0x00010000,
                         (int)(160u << 16),
                         (int)((unsigned)S << 16),
                         (int)(160u << 16),
                         AQK,
                         160, 0, 0 };
            i32x4 zz = { 0, 0, 0, 0 };
#if __has_include(<hip/amd_detail/amd_gfx1250_TDM.h>)
            i32x8 z8 = { 0, 0, 0, 0, 0, 0, 0, 0 };
            __builtin_amdgcn_tensor_load_to_lds(g0, g1, zz, zz, z8, 0);
#else
            __builtin_amdgcn_tensor_load_to_lds(g0, g1, zz, zz, 0);
#endif
        }
        wait_tensor0();
#else
        for (int it = tid; it < AQK * 20; it += 128) {
            int r = it / 20, c = (it % 20) * 8;
            *(uint4*)&Ks[r][c] = *(const uint4*)&Kg[baseQK + (size_t)(k0 + r) * 160 + c];
        }
#endif
        // ---- stage V tile transposed: Vs[d][key] (manual; needs transpose) ----
        for (int it = tid; it < AQK * 16; it += 128) {
            int r = it >> 4, c = (it & 15) * 8;
            uint4 t = *(const uint4*)&V[baseV + (size_t)(k0 + r) * 128 + c];
            const __bf16* tb = (const __bf16*)&t;
#pragma unroll
            for (int e = 0; e < 8; ++e) Vs[c + e][r] = tb[e];
        }
        __syncthreads();

#pragma unroll
        for (int kc = 0; kc < AQK; kc += 32) {
            // ---- scores: 16 q-rows x 32 keys ----
            v8f sc[2];
            sc[0] = (v8f)(0.0f); sc[1] = (v8f)(0.0f);
#pragma unroll
            for (int ni = 0; ni < 2; ++ni) {
                const int key = kc + ni * 16 + lrow;
                Frag16 kf[5];
#pragma unroll
                for (int d0 = 0; d0 < 5; ++d0) {
                    const __bf16* kp = &Ks[key][d0 * 32 + kb8];
                    kf[d0].q[0] = *(const uint4*)(kp);
                    kf[d0].q[1] = *(const uint4*)(kp + 16);
                }
#pragma unroll
                for (int d0 = 0; d0 < 5; ++d0)
                    sc[ni] = __builtin_amdgcn_wmma_f32_16x16x32_bf16(
                        false, qf[d0].v, false, kf[d0].v, (short)0, sc[ni],
                        false, false);
            }
            // ---- online softmax ----
            float mnew[8], rs[8];
#pragma unroll
            for (int r = 0; r < 8; ++r) {
                float t = fmaxf(sc[0][r], sc[1][r]) * scale;
#pragma unroll
                for (int m = 1; m < 16; m <<= 1)
                    t = fmaxf(t, __shfl_xor(t, m, 32));
                mnew[r] = fmaxf(mrow[r], t);
            }
            float rescale[8];
#pragma unroll
            for (int r = 0; r < 8; ++r)
                rescale[r] = exp2f((mrow[r] - mnew[r]) * LOG2E);
#pragma unroll
            for (int j = 0; j < 8; ++j)
#pragma unroll
                for (int r = 0; r < 8; ++r) acc[j][r] *= rescale[r];
#pragma unroll
            for (int r = 0; r < 8; ++r) rs[r] = 0.0f;
#pragma unroll
            for (int ni = 0; ni < 2; ++ni)
#pragma unroll
                for (int r = 0; r < 8; ++r) {
                    float p = exp2f((sc[ni][r] * scale - mnew[r]) * LOG2E);
                    sc[ni][r] = p;
                    rs[r] += p;
                }
#pragma unroll
            for (int r = 0; r < 8; ++r) {
                float t = rs[r];
#pragma unroll
                for (int m = 1; m < 16; m <<= 1)
                    t += __shfl_xor(t, m, 32);
                lsum[r] = lsum[r] * rescale[r] + t;
                mrow[r] = mnew[r];
            }
            // ---- P: C-layout f32 -> LDS -> A-layout bf16 ----
#pragma unroll
            for (int ni = 0; ni < 2; ++ni)
#pragma unroll
                for (int r = 0; r < 8; ++r)
                    Ps[wave][r + 8 * hi][ni * 16 + lrow] = f2bf(sc[ni][r]);
            Frag16 pf;  // per-wave LDS, same-wave DS ordering guarantees RAW
            pf.q[0] = *(const uint4*)&Ps[wave][lrow][kb8];
            pf.q[1] = *(const uint4*)&Ps[wave][lrow][kb8 + 16];
            // ---- acc += P @ V (K=32 keys, N=128) ----
#pragma unroll
            for (int j = 0; j < 8; ++j) {
                Frag16 vf;
                const __bf16* vp = &Vs[j * 16 + lrow][kc + kb8];
                vf.q[0] = *(const uint4*)(vp);
                vf.q[1] = *(const uint4*)(vp + 16);
                acc[j] = __builtin_amdgcn_wmma_f32_16x16x32_bf16(
                    false, pf.v, false, vf.v, (short)0, acc[j], false, false);
            }
        }
        __syncthreads();
    }

    // ---- normalize + store O[b*S + row][h*128 + d] ----
    float inv[8];
#pragma unroll
    for (int r = 0; r < 8; ++r) inv[r] = 1.0f / lsum[r];
#pragma unroll
    for (int j = 0; j < 8; ++j) {
        int col = h * 128 + j * 16 + lrow;
#pragma unroll
        for (int r = 0; r < 8; ++r) {
            size_t grow = (size_t)b * S + q0 + wave * 16 + r + 8 * hi;
            O[grow * 2048 + col] = f2bf(acc[j][r] * inv[r]);
        }
    }
}

// ---------------------------------------------------------------------------
// Host-side launch
// ---------------------------------------------------------------------------
extern "C" void kernel_launch(void* const* d_in, const int* in_sizes, int n_in,
                              void* d_out, int out_size, void* d_ws, size_t ws_size,
                              hipStream_t stream) {
    (void)in_sizes; (void)n_in; (void)out_size; (void)ws_size;
    const int B = 4, S = 2048, NH = 16;
    const int QK = 2560, H = 2048;
    const size_t MS = (size_t)B * S;               // 8192 rows

    const float* hid  = (const float*)d_in[0];
    const float* cond = (const float*)d_in[1];
    const float* cs   = (const float*)d_in[2];
    const float* sn   = (const float*)d_in[3];
    const float* Wq   = (const float*)d_in[4];
    const float* Wk   = (const float*)d_in[5];
    const float* Wv   = (const float*)d_in[6];
    const float* Wo   = (const float*)d_in[7];
    float* out = (float*)d_out;

    char* ws = (char*)d_ws;
    size_t off = 0;
    auto carve = [&](size_t bytes) -> char* {
        char* p = ws + off;
        off += (bytes + 255) & ~(size_t)255;
        return p;
    };
    __bf16* qkin = (__bf16*)carve(MS * QK * 2);          // 42 MB
    __bf16* xh   = (__bf16*)carve(MS * H * 2);           // 34 MB
    __bf16* Wqb  = (__bf16*)carve((size_t)QK * QK * 2);  // 13 MB
    __bf16* Wkb  = (__bf16*)carve((size_t)QK * QK * 2);  // 13 MB
    __bf16* Wvb  = (__bf16*)carve((size_t)H * H * 2);    // 8 MB
    __bf16* Wob  = (__bf16*)carve((size_t)H * H * 2);    // 8 MB
    float*  tmp  = (float*)carve(MS * QK * 4);           // 84 MB (reused)
    __bf16* qbf  = (__bf16*)carve(MS * QK * 2);          // 42 MB head-major
    __bf16* kbf  = (__bf16*)carve(MS * QK * 2);          // 42 MB head-major
    __bf16* vbf  = (__bf16*)carve(MS * H * 2);           // 34 MB head-major
    __bf16* abf  = (__bf16*)carve(MS * H * 2);           // 34 MB attn out

    // 1) pack inputs to bf16
    {
        size_t total = MS * QK;
        pack_inputs<<<dim3((unsigned)((total + 255) / 256)), 256, 0, stream>>>(
            hid, cond, qkin, xh, total);
    }
    // 2) weights to bf16
    {
        size_t nq = (size_t)QK * QK, nh = (size_t)H * H;
        conv_f32_bf16<<<dim3((unsigned)((nq + 255) / 256)), 256, 0, stream>>>(Wq, Wqb, nq);
        conv_f32_bf16<<<dim3((unsigned)((nq + 255) / 256)), 256, 0, stream>>>(Wk, Wkb, nq);
        conv_f32_bf16<<<dim3((unsigned)((nh + 255) / 256)), 256, 0, stream>>>(Wv, Wvb, nh);
        conv_f32_bf16<<<dim3((unsigned)((nh + 255) / 256)), 256, 0, stream>>>(Wo, Wob, nh);
    }
    // 3) Q = qkin @ Wq^T  -> rope -> qbf
    gemm_bf16_f32<<<dim3(QK / GBN, (unsigned)(MS / GBM)), 256, 0, stream>>>(
        qkin, Wqb, tmp, (int)MS, QK, QK);
    {
        size_t total = MS * QK;
        rope_repack<<<dim3((unsigned)((total + 255) / 256)), 256, 0, stream>>>(
            tmp, cs, sn, qbf, S, total);
    }
    // 4) K = qkin @ Wk^T  -> rope -> kbf
    gemm_bf16_f32<<<dim3(QK / GBN, (unsigned)(MS / GBM)), 256, 0, stream>>>(
        qkin, Wkb, tmp, (int)MS, QK, QK);
    {
        size_t total = MS * QK;
        rope_repack<<<dim3((unsigned)((total + 255) / 256)), 256, 0, stream>>>(
            tmp, cs, sn, kbf, S, total);
    }
    // 5) V = hidden @ Wv^T -> head-major bf16
    gemm_bf16_f32<<<dim3(H / GBN, (unsigned)(MS / GBM)), 256, 0, stream>>>(
        xh, Wvb, tmp, (int)MS, H, H);
    {
        size_t total = MS * H;
        v_repack<<<dim3((unsigned)((total + 255) / 256)), 256, 0, stream>>>(
            tmp, vbf, S, total);
    }
    // 6) flash attention -> abf [B*S, 2048]
    attn_flash<<<dim3(S / 64, NH, B), 128, 0, stream>>>(qbf, kbf, vbf, abf, S);
    // 7) out = attn @ Wo^T (fp32 straight into d_out)
    gemm_bf16_f32<<<dim3(H / GBN, (unsigned)(MS / GBM)), 256, 0, stream>>>(
        abf, Wob, out, (int)MS, H, H);
}
